// ShiftedResNet_14345190769020
// MI455X (gfx1250) — matrix operand order repaired and checked by reading, not compiled, not run on hardware
//
#include <hip/hip_runtime.h>
#include <stdint.h>
#include <stddef.h>

// ---------------- problem constants (from reference) ----------------
#define DEPTH   100
#define HIDDEN  1024
#define BATCH   4096
#define NIN     512
#define NOUT    512
#define SHIFTN  25
#define LIN     0.97f

// ---------------- types ----------------
typedef __attribute__((ext_vector_type(16))) __bf16 v16bf;
typedef __attribute__((ext_vector_type(8)))  float  v8f;
typedef __attribute__((ext_vector_type(4)))  int    v4i;

#define GAS __attribute__((address_space(1)))
#define LAS __attribute__((address_space(3)))

// ---------------- CDNA5 async-copy availability probes ----------------
#if defined(__has_builtin)
#if __has_builtin(__builtin_amdgcn_global_load_async_to_lds_b128)
#define HAVE_ASYNC 1
#endif
#if __has_builtin(__builtin_amdgcn_s_wait_asynccnt)
#define HAVE_WAITASYNC 1
#endif
#endif
#ifndef HAVE_ASYNC
#define HAVE_ASYNC 0
#endif
#ifndef HAVE_WAITASYNC
#define HAVE_WAITASYNC 0
#endif

// ---------------- GEMM tiling ----------------
#define BM 128
#define BN 128
#define BK 32
#define NTHREADS 256   // 8 waves (wave32)

union FragU { v16bf v; uint4 q[2]; };

__device__ __forceinline__ void g2l_b128(const __bf16* g, __bf16* l) {
#if HAVE_ASYNC
  // async DMA global -> LDS, tracked by ASYNCcnt
  __builtin_amdgcn_global_load_async_to_lds_b128((GAS v4i*)(g), (LAS v4i*)(l), 0, 0);
#else
  *(uint4*)l = *(const uint4*)g;
#endif
}

__device__ __forceinline__ void wait_async_keep4() {
#if HAVE_ASYNC
#if HAVE_WAITASYNC
  __builtin_amdgcn_s_wait_asynccnt(4);   // leave the 4 just-issued next-stage copies in flight
#else
  asm volatile("s_wait_asynccnt 4" ::: "memory");
#endif
#endif
}

__device__ __forceinline__ void wait_async_all() {
#if HAVE_ASYNC
#if HAVE_WAITASYNC
  __builtin_amdgcn_s_wait_asynccnt(0);
#else
  asm volatile("s_wait_asynccnt 0" ::: "memory");
#endif
#endif
}

// C = op(A[M,K] (bf16) x W[N,K]^T (bf16) + bias)
// OUTMODE 0: store bf16 (optionally ReLU)         (hidden activations)
// OUTMODE 1: store f32                             (fc_in / fc_out)
// OUTMODE 2: store f32 = LIN*R + (1-LIN)*(acc+b)   (residual blend)
template <bool RELU, int OUTMODE>
__global__ __launch_bounds__(NTHREADS)
void wmma_gemm_kernel(const __bf16* __restrict__ A,
                      const __bf16* __restrict__ Bw,
                      const float*  __restrict__ bias,
                      const float*  __restrict__ Rres,
                      void* __restrict__ Cout,
                      int M, int N, int K)
{
  __shared__ __align__(16) __bf16 As[2][BM * BK];  // [row][k]
  __shared__ __align__(16) __bf16 Bs[2][BN * BK];  // [n][k]

  const int tid  = threadIdx.x;
  const int lane = tid & 31;
  const int wave = tid >> 5;
  const int wm   = (wave & 3) << 5;   // wave M offset: 0/32/64/96
  const int wn   = (wave >> 2) << 6;  // wave N offset: 0/64
  const int rowA0 = blockIdx.y * BM;
  const int rowB0 = blockIdx.x * BN;

  // Per-thread copy chunks: 512 x 16B chunks per 128x32 tile, 2 per thread per tile.
  const int c0 = tid, c1 = tid + NTHREADS;
  const int r0 = c0 >> 2, kc0 = (c0 & 3) << 3;
  const int r1 = c1 >> 2, kc1 = (c1 & 3) << 3;

  v8f acc[2][4] = {};

  const int nK = K >> 5;

  auto issue_stage = [&](int kt, int buf) {
    const int k0 = kt << 5;
    g2l_b128(A  + (size_t)(rowA0 + r0) * K + k0 + kc0, &As[buf][r0 * BK + kc0]);
    g2l_b128(Bw + (size_t)(rowB0 + r0) * K + k0 + kc0, &Bs[buf][r0 * BK + kc0]);
    g2l_b128(A  + (size_t)(rowA0 + r1) * K + k0 + kc1, &As[buf][r1 * BK + kc1]);
    g2l_b128(Bw + (size_t)(rowB0 + r1) * K + k0 + kc1, &Bs[buf][r1 * BK + kc1]);
  };

  issue_stage(0, 0);

  for (int kt = 0; kt < nK; ++kt) {
    const int cur = kt & 1;
    if (kt + 1 < nK) { issue_stage(kt + 1, cur ^ 1); wait_async_keep4(); }
    else             { wait_async_all(); }
    __syncthreads();

    // A fragment: 16-bit A 16x32 layout — lanes 0-15: K0..7 (v0-3) + K16..23 (v4-7);
    // lanes 16-31: K8..15 + K24..31. Two ds_load_b128 each.
    FragU a[2], b[4];
#pragma unroll
    for (int t = 0; t < 2; ++t) {
      const int row = wm + t * 16 + (lane & 15);
      const int ka  = (lane >> 4) << 3;          // 0 or 8
      a[t].q[0] = *(const uint4*)&As[cur][row * BK + ka];
      a[t].q[1] = *(const uint4*)&As[cur][row * BK + ka + 16];
    }
    // B fragment: 32x16, lane = column n, lanes 0-15 hold K0..7 + K16..23, lanes 16-31 K8..15 + K24..31,
    // matching the A layout => two b128 loads of W row n.
#pragma unroll
    for (int t = 0; t < 4; ++t) {
      const int nrow = wn + t * 16 + (lane & 15);
      const int kb   = (lane >> 4) << 3;         // 0 or 8
      b[t].q[0] = *(const uint4*)&Bs[cur][nrow * BK + kb];
      b[t].q[1] = *(const uint4*)&Bs[cur][nrow * BK + kb + 16];
    }

#pragma unroll
    for (int tm = 0; tm < 2; ++tm)
#pragma unroll
      for (int tn = 0; tn < 4; ++tn)
        acc[tm][tn] = __builtin_amdgcn_wmma_f32_16x16x32_bf16(
            false, a[tm].v, false, b[tn].v, (short)0, acc[tm][tn], false, false);

    __syncthreads();
  }

  // Epilogue. C/D layout: VGPR v -> M = v + 8*(lane>=16); lane&15 -> N.
#pragma unroll
  for (int tn = 0; tn < 4; ++tn) {
    const int nG = rowB0 + wn + tn * 16 + (lane & 15);
    const float bv = bias[nG];
#pragma unroll
    for (int tm = 0; tm < 2; ++tm) {
#pragma unroll
      for (int v = 0; v < 8; ++v) {
        const int mG = rowA0 + wm + tm * 16 + v + ((lane >> 4) << 3);
        float val = acc[tm][tn][v] + bv;
        if (RELU) val = fmaxf(val, 0.0f);
        const size_t idx = (size_t)mG * N + nG;
        if (OUTMODE == 0)      ((__bf16*)Cout)[idx] = (__bf16)val;
        else if (OUTMODE == 1) ((float*)Cout)[idx]  = val;
        else ((float*)Cout)[idx] = LIN * Rres[idx] + (1.0f - LIN) * val;
      }
    }
  }
}

// ---------------- helper kernels ----------------
__global__ void f32_to_bf16_kernel(const float* __restrict__ in,
                                   __bf16* __restrict__ out, int n) {
  int i = blockIdx.x * blockDim.x + threadIdx.x;
  if (i < n) out[i] = (__bf16)in[i];
}

// R = shift(H) (25 zeros prepended per row, last 25 dropped); also emit bf16 copy.
__global__ void shift_kernel(const float* __restrict__ H,
                             float* __restrict__ R,
                             __bf16* __restrict__ Rbf, int n) {
  int i = blockIdx.x * blockDim.x + threadIdx.x;
  if (i < n) {
    int col = i & (HIDDEN - 1);
    float v = (col < SHIFTN) ? 0.0f : H[i - SHIFTN];
    R[i]   = v;
    Rbf[i] = (__bf16)v;
  }
}

// ---------------- host launcher ----------------
extern "C" void kernel_launch(void* const* d_in, const int* in_sizes, int n_in,
                              void* d_out, int out_size, void* d_ws, size_t ws_size,
                              hipStream_t stream)
{
  (void)in_sizes; (void)n_in; (void)out_size; (void)ws_size;

  const float* x     = (const float*)d_in[0];
  const float* fc1_w = (const float*)d_in[1];
  const float* fc1_b = (const float*)d_in[2];
  const float* W1    = (const float*)d_in[3];
  const float* b1    = (const float*)d_in[4];
  const float* W2    = (const float*)d_in[5];
  const float* b2    = (const float*)d_in[6];
  const float* fc2_w = (const float*)d_in[7];
  const float* fc2_b = (const float*)d_in[8];

  // Workspace layout (~58 MB total):
  char* ws = (char*)d_ws;
  const size_t MB = (size_t)1 << 20;
  float*  H    = (float*) (ws + 0 * MB);   // 16 MB  [4096,1024] f32 activation
  float*  R    = (float*) (ws + 16 * MB);  // 16 MB  shifted residual f32
  __bf16* Rbf  = (__bf16*)(ws + 32 * MB);  //  8 MB  bf16 GEMM operand (reused as Hbf)
  __bf16* Abuf = (__bf16*)(ws + 40 * MB);  //  8 MB  relu(R @ W1^T + b1) activations bf16
  __bf16* xbf  = (__bf16*)(ws + 48 * MB);  //  4 MB
  __bf16* w1bf = (__bf16*)(ws + 52 * MB);  //  1 MB  fc1_w bf16
  __bf16* w2bf = (__bf16*)(ws + 53 * MB);  //  1 MB  fc2_w bf16
  __bf16* Wst1 = (__bf16*)(ws + 54 * MB);  //  2 MB  per-block W1 staging
  __bf16* Wst2 = (__bf16*)(ws + 56 * MB);  //  2 MB  per-block W2 staging

  auto cvt = [&](const float* src, __bf16* dst, int n) {
    f32_to_bf16_kernel<<<(n + 255) / 256, 256, 0, stream>>>(src, dst, n);
  };

  cvt(x,     xbf,  BATCH * NIN);
  cvt(fc1_w, w1bf, HIDDEN * NIN);
  cvt(fc2_w, w2bf, NOUT * HIDDEN);

  dim3 blk(NTHREADS);
  const dim3 gridH(HIDDEN / BN, BATCH / BM);  // N=1024 GEMMs
  const dim3 gridO(NOUT   / BN, BATCH / BM);  // N=512 final GEMM

  // H = x @ fc1_w^T + fc1_b  (f32 out)
  wmma_gemm_kernel<false, 1><<<gridH, blk, 0, stream>>>(
      xbf, w1bf, fc1_b, nullptr, H, BATCH, HIDDEN, NIN);

  const int actN = BATCH * HIDDEN;
  for (int d = 0; d < DEPTH; ++d) {
    shift_kernel<<<(actN + 255) / 256, 256, 0, stream>>>(H, R, Rbf, actN);

    cvt(W1 + (size_t)d * HIDDEN * HIDDEN, Wst1, HIDDEN * HIDDEN);
    // A2 = relu(R @ W1^T + b1)  (bf16 out)
    wmma_gemm_kernel<true, 0><<<gridH, blk, 0, stream>>>(
        Rbf, Wst1, b1 + (size_t)d * HIDDEN, nullptr, Abuf, BATCH, HIDDEN, HIDDEN);

    cvt(W2 + (size_t)d * HIDDEN * HIDDEN, Wst2, HIDDEN * HIDDEN);
    // H = 0.97*R + 0.03*(A2 @ W2^T + b2)  (f32 out)
    wmma_gemm_kernel<false, 2><<<gridH, blk, 0, stream>>>(
        Abuf, Wst2, b2 + (size_t)d * HIDDEN, R, H, BATCH, HIDDEN, HIDDEN);
  }

  // out = H @ fc2_w^T + fc2_b  (f32 out)
  cvt(H, Rbf, actN);  // Hbf (reuse Rbf buffer)
  wmma_gemm_kernel<false, 1><<<gridO, blk, 0, stream>>>(
      Rbf, w2bf, fc2_b, nullptr, d_out, BATCH, NOUT, HIDDEN);
}
